// RNNLSTMBinaryClassifier_74062416052799
// MI455X (gfx1250) — matrix operand order repaired and checked by reading, not compiled, not run on hardware
//
#include <hip/hip_runtime.h>
#include <hip/hip_bf16.h>
#include <cstdint>

// ---------------- problem constants ----------------
#define BATCH 64
#define TSTEPS 512
#define DIN 2048
#define HID 128
#define GATES 512               // 4*HID
#define BTILE 16                // batch rows per recurrent workgroup
#define MROWS (BATCH * TSTEPS)  // 32768 GEMM rows

// ---------------- WMMA vector types ----------------
typedef __attribute__((ext_vector_type(16))) __bf16 v16bf;
typedef __attribute__((ext_vector_type(8)))  float  v8f;
typedef __attribute__((ext_vector_type(4)))  __bf16 v4bf;

__device__ __forceinline__ float sigmoidf_fast(float x) {
    return 1.0f / (1.0f + __expf(-x));
}

// =====================================================================
// fp32 -> bf16 weight conversion (once, outside the hot loops)
// =====================================================================
__global__ void cvt_f32_bf16_kernel(const float* __restrict__ src,
                                    __bf16* __restrict__ dst, int n)
{
    const int i = (blockIdx.x * blockDim.x + threadIdx.x) * 4;
    if (i < n) {
        const float4 v = *(const float4*)(src + i);
        v4bf o;
        o[0] = (__bf16)v.x; o[1] = (__bf16)v.y;
        o[2] = (__bf16)v.z; o[3] = (__bf16)v.w;
        *(v4bf*)(dst + i) = o;
    }
}

// =====================================================================
// Layer-0 GEMM: C[M,512] = x[M,2048] @ Wb^T + (b1+b2), bf16 WMMA, f32 acc.
// Block = 256 threads (8 waves) owns 16 M rows and all N=512.
// A is staged fp32 -> LDS via GLOBAL_LOAD_ASYNC_TO_LDS_B128 (double
// buffered, ASYNCcnt-tracked) so the 256MB x stream is read exactly once
// per block instead of once per wave. Waves convert to bf16 on LDS read.
// =====================================================================
#define KCHUNK 128
#define SA_STRIDE 132   // fp32 elems per LDS row (128 + pad), 528B, 16B-aligned

__global__ void gemm_x_async_kernel(const float* __restrict__ A, int K,
                                    const __bf16* __restrict__ Wb,
                                    const float* __restrict__ b1,
                                    const float* __restrict__ b2,
                                    float* __restrict__ C)
{
    __shared__ float sA[2][16][SA_STRIDE];

    const int tid   = threadIdx.x;
    const int lane  = tid & 31;
    const int wave  = tid >> 5;
    const int mBase = blockIdx.x * 16;
    const int mr    = lane & 15;
    const int half  = lane >> 4;
    const int kOffA = half * 16;
    const int kOffB = half * 16;

    v8f acc[4] = {};

    // 16 rows x 32 16B-segments = 512 segments; 2 per thread
    const int r0 = tid >> 5,         c0 = (tid & 31) * 4;
    const int r1 = (tid + 256) >> 5, c1 = (tid & 31) * 4;

    auto issue_chunk = [&](int buf, int kc) {
        const float* g0 = A + (size_t)(mBase + r0) * K + kc + c0;
        const float* g1 = A + (size_t)(mBase + r1) * K + kc + c1;
        const unsigned l0 = (unsigned)(uintptr_t)&sA[buf][r0][c0];
        const unsigned l1 = (unsigned)(uintptr_t)&sA[buf][r1][c1];
        asm volatile("global_load_async_to_lds_b128 %0, %2, off\n\t"
                     "global_load_async_to_lds_b128 %1, %3, off"
                     :: "v"(l0), "v"(l1), "v"(g0), "v"(g1)
                     : "memory");
    };

    const int nChunks = K / KCHUNK;
    issue_chunk(0, 0);

    for (int c = 0; c < nChunks; ++c) {
        const int buf = c & 1;
        if (c + 1 < nChunks) {
            issue_chunk(buf ^ 1, (c + 1) * KCHUNK);
            asm volatile("s_wait_asynccnt 0x2" ::: "memory"); // chunk c landed
        } else {
            asm volatile("s_wait_asynccnt 0x0" ::: "memory");
        }
        __syncthreads();   // all waves' segments of chunk c visible

        const int kcBase = c * KCHUNK;
#pragma unroll
        for (int kk = 0; kk < 4; ++kk) {           // 4 x K32 per chunk
            const int k0 = kk * 32;
            v16bf a;
#pragma unroll
            for (int j = 0; j < 8; ++j) {
                a[j]     = (__bf16)sA[buf][mr][k0 + kOffA + j];
                a[8 + j] = (__bf16)sA[buf][mr][k0 + 8 + kOffA + j];
            }
#pragma unroll
            for (int nt = 0; nt < 4; ++nt) {
                const int n = wave * 64 + nt * 16 + mr;
                const __bf16* __restrict__ wrow =
                    Wb + (size_t)n * K + kcBase + k0 + kOffB;
                v16bf b;
#pragma unroll
                for (int j = 0; j < 16; ++j) b[j] = wrow[j];
                acc[nt] = __builtin_amdgcn_wmma_f32_16x16x32_bf16(
                    false, a, false, b, (short)0, acc[nt], false, false);
            }
        }
        __syncthreads();   // done reading buf before it is refilled
    }

#pragma unroll
    for (int nt = 0; nt < 4; ++nt) {
        const int n    = wave * 64 + nt * 16 + mr;
        const float bs = b1[n] + b2[n];
#pragma unroll
        for (int r = 0; r < 8; ++r) {
            const int m = mBase + r + half * 8;
            C[(size_t)m * GATES + n] = acc[nt][r] + bs;
        }
    }
}

// =====================================================================
// Layer-1 GEMM: C[M,512] = out0[M,128] @ Wb1^T + (b1+b2). A already bf16
// and only 8MB, so direct per-wave loads are fine.
// =====================================================================
__global__ void gemm_direct_kernel(const __bf16* __restrict__ A, int K,
                                   const __bf16* __restrict__ Wb,
                                   const float* __restrict__ b1,
                                   const float* __restrict__ b2,
                                   float* __restrict__ C)
{
    const int tid   = threadIdx.x;
    const int lane  = tid & 31;
    const int wave  = tid >> 5;
    const int mBase = blockIdx.x * 16;
    const int mr    = lane & 15;
    const int half  = lane >> 4;
    const int kOffA = half * 16;
    const int kOffB = half * 16;

    v8f acc[4] = {};
    const __bf16* __restrict__ arow = A + (size_t)(mBase + mr) * K;

    for (int k0 = 0; k0 < K; k0 += 32) {
        v16bf a;
#pragma unroll
        for (int j = 0; j < 8; ++j) {
            a[j]     = arow[k0 + kOffA + j];
            a[8 + j] = arow[k0 + 8 + kOffA + j];
        }
        if (k0 + 32 < K)
            __builtin_prefetch(&arow[k0 + 32 + kOffA], 0, 0);
#pragma unroll
        for (int nt = 0; nt < 4; ++nt) {
            const int n = wave * 64 + nt * 16 + mr;
            const __bf16* __restrict__ wrow = Wb + (size_t)n * K + k0 + kOffB;
            v16bf b;
#pragma unroll
            for (int j = 0; j < 16; ++j) b[j] = wrow[j];
            acc[nt] = __builtin_amdgcn_wmma_f32_16x16x32_bf16(
                false, a, false, b, (short)0, acc[nt], false, false);
        }
    }

#pragma unroll
    for (int nt = 0; nt < 4; ++nt) {
        const int n    = wave * 64 + nt * 16 + mr;
        const float bs = b1[n] + b2[n];
#pragma unroll
        for (int r = 0; r < 8; ++r) {
            const int m = mBase + r + half * 8;
            C[(size_t)m * GATES + n] = acc[nt][r] + bs;
        }
    }
}

// =====================================================================
// LSTM recurrence (unchanged from round 1 — wmma/ds/trans mix verified).
// =====================================================================
#define SW_STRIDE 136
#define SH_STRIDE 132
#define SW_BYTES  (GATES * SW_STRIDE * 2)
#define SH_BYTES  (BTILE * SH_STRIDE * 4)
#define SG_BYTES  (BTILE * GATES * 4)
#define LDS_TOTAL (SW_BYTES + SH_BYTES + SG_BYTES)   // 180480 < 320KB

template <bool STORE_SEQ>
__global__ void lstm_rec_kernel(const float* __restrict__ xproj, // [B*T, 512]
                                const float* __restrict__ Whh,   // [512, 128] fp32
                                __bf16* __restrict__ outSeq,     // [B*T, 128] bf16
                                float* __restrict__ hFinal,      // [B, 128] fp32
                                int T)
{
    extern __shared__ char smem[];
    __bf16* sW = (__bf16*)smem;
    float*  sH = (float*)(smem + SW_BYTES);
    float*  sG = (float*)(smem + SW_BYTES + SH_BYTES);

    const int tid   = threadIdx.x;
    const int lane  = tid & 31;
    const int wave  = tid >> 5;
    const int bBase = blockIdx.x * BTILE;
    const int mr    = lane & 15;
    const int half  = lane >> 4;
    const int kOffA = half * 16;
    const int kOffB = half * 16;
    const int gate  = wave >> 1;

    for (int i = tid; i < GATES * HID; i += 256) {
        const int n = i >> 7, k = i & 127;
        sW[n * SW_STRIDE + k] = (__bf16)Whh[i];
    }
    for (int i = tid; i < BTILE * SH_STRIDE; i += 256) sH[i] = 0.0f;

    float c[8];
#pragma unroll
    for (int j = 0; j < 8; ++j) c[j] = 0.0f;
    __syncthreads();

    for (int t = 0; t < T; ++t) {
#pragma unroll
        for (int nt = 0; nt < 4; ++nt) {
            const int n0 = wave * 64 + nt * 16;
            v8f acc = {};
#pragma unroll
            for (int kk = 0; kk < 4; ++kk) {
                const int k0 = kk * 32;
                v16bf a, b;
#pragma unroll
                for (int j = 0; j < 8; ++j) {
                    a[j]     = (__bf16)sH[mr * SH_STRIDE + k0 + kOffA + j];
                    a[8 + j] = (__bf16)sH[mr * SH_STRIDE + k0 + 8 + kOffA + j];
                }
                const __bf16* wrow = &sW[(n0 + mr) * SW_STRIDE + k0 + kOffB];
#pragma unroll
                for (int j = 0; j < 16; ++j) b[j] = wrow[j];
                acc = __builtin_amdgcn_wmma_f32_16x16x32_bf16(
                    false, a, false, b, (short)0, acc, false, false);
            }
            const int n = n0 + mr;
#pragma unroll
            for (int r = 0; r < 8; ++r) {
                const int row = r + half * 8;
                const long grow = (long)(bBase + row) * T + t;
                const float v = acc[r] + xproj[grow * GATES + n];
                const float av = (gate == 2) ? tanhf(v) : sigmoidf_fast(v);
                sG[row * GATES + n] = av;
            }
        }
        __syncthreads();

#pragma unroll
        for (int j = 0; j < 8; ++j) {
            const int e   = tid * 8 + j;
            const int row = e >> 7;
            const int hh  = e & 127;
            const float iv = sG[row * GATES + hh];
            const float fv = sG[row * GATES + 128 + hh];
            const float gv = sG[row * GATES + 256 + hh];
            const float ov = sG[row * GATES + 384 + hh];
            c[j] = fv * c[j] + iv * gv;
            const float hv = ov * tanhf(c[j]);
            sH[row * SH_STRIDE + hh] = hv;
            if (STORE_SEQ) {
                outSeq[((long)(bBase + row) * T + t) * HID + hh] = (__bf16)hv;
            } else if (t == T - 1) {
                hFinal[(bBase + row) * HID + hh] = hv;
            }
        }
        __syncthreads();
    }
}

// =====================================================================
// Final FC: out[b] = dot(h1[b,:], Wfc[0,:]) + bfc
// =====================================================================
__global__ void fc_kernel(const float* __restrict__ h1,
                          const float* __restrict__ Wfc,
                          const float* __restrict__ bfc,
                          float* __restrict__ out)
{
    __shared__ float red[HID];
    const int b = blockIdx.x, tid = threadIdx.x;
    red[tid] = h1[b * HID + tid] * Wfc[tid];
    __syncthreads();
#pragma unroll
    for (int s = 64; s > 0; s >>= 1) {
        if (tid < s) red[tid] += red[tid + s];
        __syncthreads();
    }
    if (tid == 0) out[b] = red[0] + bfc[0];
}

// =====================================================================
extern "C" void kernel_launch(void* const* d_in, const int* in_sizes, int n_in,
                              void* d_out, int out_size, void* d_ws, size_t ws_size,
                              hipStream_t stream)
{
    const float* x    = (const float*)d_in[0];
    const float* Wih0 = (const float*)d_in[1];
    const float* Whh0 = (const float*)d_in[2];
    const float* bih0 = (const float*)d_in[3];
    const float* bhh0 = (const float*)d_in[4];
    const float* Wih1 = (const float*)d_in[5];
    const float* Whh1 = (const float*)d_in[6];
    const float* bih1 = (const float*)d_in[7];
    const float* bhh1 = (const float*)d_in[8];
    const float* Wfc  = (const float*)d_in[9];
    const float* bfc  = (const float*)d_in[10];

    // workspace layout
    char* ws = (char*)d_ws;
    float*  xproj = (float*)ws;                                  // 64MB (both layers)
    __bf16* out0  = (__bf16*)(ws + (size_t)MROWS * GATES * 4);   // 8MB
    char*   p     = ws + (size_t)MROWS * GATES * 4 + (size_t)MROWS * HID * 2;
    float*  h1    = (float*)p;                                   // 32KB
    __bf16* Wb0   = (__bf16*)(p + 65536);                        // 2MB
    __bf16* Wb1   = (__bf16*)(p + 65536 + (size_t)GATES * DIN * 2); // 128KB

    const dim3 blk(256);

    // weight pre-conversion to bf16
    cvt_f32_bf16_kernel<<<(GATES * DIN / 4 + 255) / 256, blk, 0, stream>>>(Wih0, Wb0, GATES * DIN);
    cvt_f32_bf16_kernel<<<(GATES * HID / 4 + 255) / 256, blk, 0, stream>>>(Wih1, Wb1, GATES * HID);

    // Layer 0: xproj = x @ Wih0^T + bih0 + bhh0 (async-LDS staged A)
    gemm_x_async_kernel<<<MROWS / 16, blk, 0, stream>>>(x, DIN, Wb0, bih0, bhh0, xproj);
    // Layer 0 recurrence -> out0 (bf16 sequence)
    lstm_rec_kernel<true><<<BATCH / BTILE, blk, LDS_TOTAL, stream>>>(xproj, Whh0, out0, nullptr, TSTEPS);
    // Layer 1: xproj = out0 @ Wih1^T + bih1 + bhh1
    gemm_direct_kernel<<<MROWS / 16, blk, 0, stream>>>(out0, HID, Wb1, bih1, bhh1, xproj);
    // Layer 1 recurrence -> final h1 only
    lstm_rec_kernel<false><<<BATCH / BTILE, blk, LDS_TOTAL, stream>>>(xproj, Whh1, nullptr, h1, TSTEPS);
    // FC head
    fc_kernel<<<BATCH, HID, 0, stream>>>(h1, Wfc, bfc, (float*)d_out);
}
